// SocialPooling_19670950216138
// MI455X (gfx1250) — matrix operand order repaired and checked by reading, not compiled, not run on hardware
//
#include <hip/hip_runtime.h>

// ---------------------------------------------------------------------------
// Social pooling, fused per-scene kernel for gfx1250 (MI455X).
//   out[i,p] = bias[p] + sum_{g=0..15} sum_{j: code(i,j)==g} T[j,g,p]
//   T[j,g,p] = sum_h  h[j,h] * W[g*128+h, p]     (bf16 WMMA, f32 accumulate)
// One workgroup (256 threads = 8 wave32) per scene; all intermediates in LDS.
// ---------------------------------------------------------------------------

#define HIDDEN 128
#define POOL   64
#define GRIDSZ 4
#define NCELL  16          // GRIDSZ*GRIDSZ
#define PEDS   64
#define NBHD   2.0f

// Padded LDS strides: rows stay 16B-aligned (b128-friendly) AND rotate across
// bank groups (bank = (4*row + dword_col) % 64 -> conflict-free frag loads).
#define HSTR 136           // bf16 elems per Hb row (272 B)
#define WSTR 136           // bf16 elems per Wg row (272 B)
#define TSTR 68            // f32 elems per Tg row  (272 B)

typedef __attribute__((ext_vector_type(16))) __bf16        bf16x16;
typedef __attribute__((ext_vector_type(8)))  float         f32x8;
typedef __attribute__((ext_vector_type(4)))  float         f32x4;
typedef __attribute__((ext_vector_type(4)))  unsigned int  u32x4;
typedef __attribute__((ext_vector_type(2)))  unsigned int  u32x2;

union Frag {
    bf16x16 v;
    u32x4   q[2];
};

static __device__ __forceinline__ unsigned int f32_to_bf16_rne(float f) {
    unsigned int u = __builtin_bit_cast(unsigned int, f);
    unsigned int r = u + 0x7fffu + ((u >> 16) & 1u);   // round-to-nearest-even
    return r >> 16;
}

__global__ __launch_bounds__(256)
void social_pool_fused(const float* __restrict__ hidden,
                       const float* __restrict__ pos,
                       const float* __restrict__ W,      // [NCELL*HIDDEN, POOL]
                       const float* __restrict__ bias,   // [POOL]
                       float* __restrict__ out)          // [scenes*PEDS, POOL]
{
    __shared__ __align__(16) unsigned short sHb[PEDS * HSTR];  // bf16 hidden, row j
    __shared__ __align__(16) unsigned short sWg[POOL * WSTR];  // bf16 W_g^T, row p, col h
    __shared__ __align__(16) float          sTg[PEDS * TSTR];  // f32 T_g, row j, col p
    __shared__ __align__(4)  signed char    sCode[PEDS * PEDS];
    __shared__ __align__(8)  float          sPos[PEDS * 2];

    const int scene = blockIdx.x;
    const int tid   = threadIdx.x;
    // Scalarize the wave id so tile loops get scalar control flow (WMMA wants
    // EXEC all-ones; keep the compiler from building EXEC-masked loops).
    const int wave  = __builtin_amdgcn_readfirstlane(tid >> 5);
    const int lane  = tid & 31;
    const int half  = lane >> 4;   // 0: lanes 0-15, 1: lanes 16-31
    const int lm    = lane & 15;

    // ---- Stage 0: positions + hidden states (f32 -> bf16) into LDS --------
    if (tid < PEDS * 2)
        sPos[tid] = pos[scene * PEDS * 2 + tid];

    for (int e = tid * 4; e < PEDS * HIDDEN; e += 256 * 4) {
        int j = e >> 7;            // HIDDEN == 128
        int h = e & 127;
        f32x4 x = *(const f32x4*)&hidden[(size_t)(scene * PEDS + j) * HIDDEN + h];
        u32x2 pk;
        pk.x = f32_to_bf16_rne(x.x) | (f32_to_bf16_rne(x.y) << 16);
        pk.y = f32_to_bf16_rne(x.z) | (f32_to_bf16_rne(x.w) << 16);
        *(u32x2*)&sHb[j * HSTR + h] = pk;   // ds_store_b64
    }
    __syncthreads();

    // ---- Stage 0.5: neighbor grid codes (255 = masked) ---------------------
    for (int e = tid; e < PEDS * PEDS; e += 256) {
        int i = e >> 6;
        int j = e & 63;
        float dx = sPos[2 * j]     - sPos[2 * i];
        float dy = sPos[2 * j + 1] - sPos[2 * i + 1];
        signed char c = -1;
        if (i != j && fabsf(dx) <= NBHD && fabsf(dy) <= NBHD) {
            // scale = GRIDSZ / (2*NBHD) == 1
            int gx = (int)floorf(dx + NBHD);
            int gy = (int)floorf(dy + NBHD);
            gx = min(max(gx, 0), GRIDSZ - 1);
            gy = min(max(gy, 0), GRIDSZ - 1);
            c = (signed char)(gy * GRIDSZ + gx);
        }
        sCode[e] = c;
    }
    __syncthreads();

    // Per-thread output ownership: ped i_mine, 16 consecutive pool dims at p0.
    const int i_mine = tid >> 2;
    const int p0     = (tid & 3) * 16;

    // Preload my code row once (64 bytes = 16 dwords); reused for all 16 cells.
    unsigned int creg[16];
    {
        const unsigned int* crow32 = (const unsigned int*)&sCode[i_mine * PEDS];
        #pragma unroll
        for (int w = 0; w < 16; ++w) creg[w] = crow32[w];
    }

    f32x4 acc4[4];
    #pragma unroll
    for (int k = 0; k < 4; ++k) acc4[k] = (f32x4){0.f, 0.f, 0.f, 0.f};

    // ---- Main loop over the 16 grid cells ---------------------------------
    for (int g = 0; g < NCELL; ++g) {
        __syncthreads();  // prior Tg reads done before refilling Wg/Tg

        // Stage A: W_g slice -> LDS, transposed [p][h], bf16. Coalesced b128 reads.
        for (int e = tid * 4; e < POOL * HIDDEN; e += 256 * 4) {
            int h = e >> 6;        // POOL == 64
            int p = e & 63;        // p, p+1, p+2, p+3 (64 % 4 == 0)
            f32x4 x = *(const f32x4*)&W[(size_t)(g * HIDDEN + h) * POOL + p];
            sWg[(p + 0) * WSTR + h] = (unsigned short)f32_to_bf16_rne(x.x);
            sWg[(p + 1) * WSTR + h] = (unsigned short)f32_to_bf16_rne(x.y);
            sWg[(p + 2) * WSTR + h] = (unsigned short)f32_to_bf16_rne(x.z);
            sWg[(p + 3) * WSTR + h] = (unsigned short)f32_to_bf16_rne(x.w);
        }
        if (g + 1 < NCELL)  // warm L2/WGP$ for next cell's weight slice
            __builtin_prefetch(&W[(size_t)((g + 1) * HIDDEN + (tid & 127)) * POOL], 0, 1);
        __syncthreads();

        // Stage B: T_g[64x64] = Hb[64x128] @ W_g[128x64] via bf16 WMMA.
        // 16 output tiles of 16x16; 2 per wave (scalar indices); K = 4 x 32.
        #pragma unroll
        for (int rep = 0; rep < 2; ++rep) {
            const int tt    = wave + rep * 8;       // scalar
            const int mtile = tt >> 2;              // scalar
            const int ntile = tt & 3;               // scalar
            const unsigned short* aBase = &sHb[(mtile * 16 + lm) * HSTR + 8 * half];
            const unsigned short* bBase = &sWg[(ntile * 16 + lm) * WSTR + 8 * half];

            // Preload all 8 fragments (2x ds_load_b128 each), then 2
            // independent WMMA chains to hide the D->C RAW hazard.
            Frag a[4], b[4];
            #pragma unroll
            for (int ks = 0; ks < 4; ++ks) {
                // A 16x32 bf16: lanes 0-15 hold K {0..7,16..23}, lanes 16-31 {8..15,24..31}
                a[ks].q[0] = *(const u32x4*)(aBase + 32 * ks);
                a[ks].q[1] = *(const u32x4*)(aBase + 32 * ks + 16);
                // B 32x16 bf16: lanes 0-15 hold K {0..7,16..23}, lanes 16-31 {8..15,24..31}
                b[ks].q[0] = *(const u32x4*)(bBase + 32 * ks);
                b[ks].q[1] = *(const u32x4*)(bBase + 32 * ks + 16);
            }
            f32x8 c0 = {}, c1 = {};
            c0 = __builtin_amdgcn_wmma_f32_16x16x32_bf16(false, a[0].v, false, b[0].v, (short)0, c0, false, false);
            c1 = __builtin_amdgcn_wmma_f32_16x16x32_bf16(false, a[1].v, false, b[1].v, (short)0, c1, false, false);
            c0 = __builtin_amdgcn_wmma_f32_16x16x32_bf16(false, a[2].v, false, b[2].v, (short)0, c0, false, false);
            c1 = __builtin_amdgcn_wmma_f32_16x16x32_bf16(false, a[3].v, false, b[3].v, (short)0, c1, false, false);

            // C/D 16x16 f32 layout: VGPR r -> M=r (lanes 0-15) / M=r+8 (lanes 16-31).
            #pragma unroll
            for (int r = 0; r < 8; ++r) {
                int M = mtile * 16 + r + 8 * half;
                int N = ntile * 16 + lm;
                sTg[M * TSTR + N] = c0[r] + c1[r];
            }
        }
        __syncthreads();

        // Stage C: gather-accumulate this cell's contributions (b128 rows).
        for (int j = 0; j < PEDS; ++j) {
            int c = (creg[j >> 2] >> ((j & 3) * 8)) & 0xff;
            if (c == g) {
                const float* trow = &sTg[j * TSTR + p0];   // 16B aligned
                #pragma unroll
                for (int k = 0; k < 4; ++k)
                    acc4[k] += *(const f32x4*)(trow + 4 * k);
            }
        }
    }

    // ---- Epilogue: bias add + b128 stores ---------------------------------
    const size_t orow = (size_t)(scene * PEDS + i_mine) * POOL + p0;
    #pragma unroll
    for (int k = 0; k < 4; ++k) {
        f32x4 bb = *(const f32x4*)&bias[p0 + 4 * k];
        *(f32x4*)&out[orow + 4 * k] = acc4[k] + bb;
    }
}

extern "C" void kernel_launch(void* const* d_in, const int* in_sizes, int n_in,
                              void* d_out, int out_size, void* d_ws, size_t ws_size,
                              hipStream_t stream) {
    (void)n_in; (void)out_size; (void)d_ws; (void)ws_size;
    const float* hidden = (const float*)d_in[0];
    const float* pos    = (const float*)d_in[1];
    const float* W      = (const float*)d_in[2];
    const float* bias   = (const float*)d_in[3];
    float*       out    = (float*)d_out;

    const int total_peds = in_sizes[0] / HIDDEN;
    const int num_scenes = total_peds / PEDS;   // uniform seq_start_end

    social_pool_fused<<<num_scenes, 256, 0, stream>>>(hidden, pos, W, bias, out);
}
